// ProposalModuleRefine_19653770346944
// MI455X (gfx1250) — matrix operand order repaired and checked by reading, not compiled, not run on hardware
//
#include <hip/hip_runtime.h>

#define BATCH   32
#define NK      1024
#define NC      256
#define NPT     256      // NPOINT
#define NS      16       // NSAMPLE
#define RAD     0.3f
#define OUTCH   97
#define K0P     288      // logical layer-0 K (259 -> 288, zero padded)
#define XSP     296      // LDS stride (halfs) for 288-wide tiles (bank-conflict free)
#define YS      136      // LDS stride (halfs) for 128-wide tiles (bank-conflict free)

typedef __attribute__((ext_vector_type(16))) _Float16 v16h;
typedef __attribute__((ext_vector_type(8)))  _Float16 v8h;
typedef __attribute__((ext_vector_type(8)))  float    v8f;

__device__ __forceinline__ v8f zero8() {
  v8f z;
#pragma unroll
  for (int i = 0; i < 8; i++) z[i] = 0.f;
  return z;
}

// A fragment (16x32 f16, M x K) from an LDS row-major tile.
// ISA layout: lanes 0-15 -> M=lane, K {0..7,16..23}; lanes 16-31 -> M=lane-16, K {8..15,24..31}
__device__ __forceinline__ v16h load_a(const _Float16* base, int stride, int rbase,
                                       int kbase, int lane) {
  int row = rbase + (lane & 15);
  int k0  = kbase + ((lane & 16) ? 8 : 0);
  const _Float16* p = base + row * stride + k0;
  v8h lo = *(const v8h*)(p);
  v8h hi = *(const v8h*)(p + 16);
  v16h r;
#pragma unroll
  for (int i = 0; i < 8; i++) { r[i] = lo[i]; r[i + 8] = hi[i]; }
  return r;
}

// B fragment (32x16 f16, K x N) where B[k][n] = W[n][k], W row-major (O x stride).
// ISA layout: lanes 0-15 -> col=lane, K {0..7,16..23}; lanes 16-31 -> col=lane-16, K {8..15,24..31}
__device__ __forceinline__ v16h load_b(const _Float16* W, int stride, int obase,
                                       int kbase, int lane) {
  int o  = obase + (lane & 15);
  int k0 = kbase + ((lane & 16) ? 8 : 0);
  const _Float16* p = W + o * stride + k0;
  v8h lo = *(const v8h*)(p);
  v8h hi = *(const v8h*)(p + 16);
  v16h r;
#pragma unroll
  for (int i = 0; i < 8; i++) { r[i] = lo[i]; r[i + 8] = hi[i]; }
  return r;
}

__device__ __forceinline__ v8f wmma16(v16h a, v16h b, v8f c) {
  return __builtin_amdgcn_wmma_f32_16x16x32_f16(false, a, false, b, (short)0, c,
                                                false, false);
}

// D layout: VGPR v, lanes 0-15 -> (M=v, N=lane); lanes 16-31 -> (M=v+8, N=lane-16)
__device__ __forceinline__ void store_tile_relu(const v8f* acc, _Float16* Y, int stride,
                                                int rbase, const float* S, const float* T,
                                                int lane) {
  int col  = lane & 15;
  int rofs = (lane & 16) ? 8 : 0;
#pragma unroll
  for (int ot = 0; ot < 8; ot++) {
    int o = ot * 16 + col;
    float s = S[o], t = T[o];
#pragma unroll
    for (int v = 0; v < 8; v++) {
      float y = fmaxf(acc[ot][v] * s + t, 0.f);
      Y[(rbase + rofs + v) * stride + ot * 16 + col] = (_Float16)y;
    }
  }
}

// Stage O rows of `rowhalfs` f16 each: global(gstride) -> LDS(sstride), 2 threads/row.
__device__ __forceinline__ void stage_w(const _Float16* __restrict__ g, _Float16* s,
                                        int O, int rowhalfs, int gstride, int sstride,
                                        int tid) {
  int r = tid >> 1, half = tid & 1;
  int per = rowhalfs >> 1;  // halfs per thread, multiple of 8
  for (int rr = r; rr < O; rr += 128) {
    const uint4* src = (const uint4*)(g + rr * gstride + half * per);
    uint4* dst = (uint4*)(s + rr * sstride + half * per);
    for (int i = 0; i < (per >> 3); i++) dst[i] = src[i];
  }
}

// ---------------- weight / feature preprocessing ----------------

// sa_w0 (128 x 259) -> w0h (128 x 288) f16, channels permuted: [gf(256), gx(3), pad]
__global__ void cvt_w0_kernel(const float* __restrict__ src, _Float16* __restrict__ dst) {
  int i = blockIdx.x * 256 + threadIdx.x;
  if (i >= 128 * K0P) return;
  int o = i / K0P, k = i % K0P;
  float v = 0.f;
  if (k < 256)      v = src[o * 259 + 3 + k];
  else if (k < 259) v = src[o * 259 + (k - 256)];
  dst[i] = (_Float16)v;
}

__global__ void cvt128_kernel(const float* __restrict__ src, _Float16* __restrict__ dst) {
  int i = blockIdx.x * 256 + threadIdx.x;
  if (i < 128 * 128) dst[i] = (_Float16)src[i];
}

// w3 (97 x 128) -> (112 x 128) f16, zero-padded rows
__global__ void cvt_w3_kernel(const float* __restrict__ src, _Float16* __restrict__ dst) {
  int i = blockIdx.x * 256 + threadIdx.x;
  if (i >= 112 * 128) return;
  int o = i >> 7, k = i & 127;
  dst[i] = (o < OUTCH) ? (_Float16)src[o * 128 + k] : (_Float16)0.f;
}

// features (B, C=256, K=1024) f32 -> featT (B, K, C) f16
__global__ void feat_transpose_kernel(const float* __restrict__ f, _Float16* __restrict__ ft) {
  __shared__ float tile[32][33];
  int b = blockIdx.z;
  int i0 = blockIdx.x * 32;  // K
  int c0 = blockIdx.y * 32;  // C
  int tx = threadIdx.x, ty = threadIdx.y;
#pragma unroll
  for (int j = 0; j < 32; j += 8)
    tile[ty + j][tx] = f[((size_t)b * NC + (c0 + ty + j)) * NK + i0 + tx];
  __syncthreads();
#pragma unroll
  for (int j = 0; j < 32; j += 8)
    ft[((size_t)b * NK + (i0 + ty + j)) * NC + c0 + tx] = (_Float16)tile[tx][ty + j];
}

// ---------------- FPS: 1 block / batch, 1024 threads ----------------

__global__ __launch_bounds__(1024) void fps_kernel(const float* __restrict__ xyz,
                                                   float* __restrict__ new_xyz) {
  int b = blockIdx.x;
  const float* X = xyz + (size_t)b * NK * 3;
  int tid  = threadIdx.x;
  int lane = tid & 31, wid = tid >> 5;
  float px = X[tid * 3 + 0], py = X[tid * 3 + 1], pz = X[tid * 3 + 2];
  float dist = 1e10f;
  __shared__ float sv[32];
  __shared__ int   si[32];
  __shared__ int   sfar;
  int far = 0;
  for (int t = 0; t < NPT; t++) {
    if (tid < 3) new_xyz[((size_t)b * NPT + t) * 3 + tid] = X[far * 3 + tid];
    float cx = X[far * 3 + 0], cy = X[far * 3 + 1], cz = X[far * 3 + 2];
    float dx = px - cx, dy = py - cy, dz = pz - cz;
    // avoid fma contraction: match sum-of-squares ordering
    float d = __fadd_rn(__fadd_rn(__fmul_rn(dx, dx), __fmul_rn(dy, dy)), __fmul_rn(dz, dz));
    dist = fminf(dist, d);
    float v = dist;
    int   idx = tid;
#pragma unroll
    for (int off = 16; off > 0; off >>= 1) {
      float v2 = __shfl_xor(v, off, 32);
      int   i2 = __shfl_xor(idx, off, 32);
      if (v2 > v || (v2 == v && i2 < idx)) { v = v2; idx = i2; }  // first-max tie-break
    }
    __syncthreads();
    if (lane == 0) { sv[wid] = v; si[wid] = idx; }
    __syncthreads();
    if (wid == 0) {
      float vv = sv[lane];
      int   ii = si[lane];
#pragma unroll
      for (int off = 16; off > 0; off >>= 1) {
        float v2 = __shfl_xor(vv, off, 32);
        int   i2 = __shfl_xor(ii, off, 32);
        if (v2 > vv || (v2 == vv && i2 < ii)) { vv = v2; ii = i2; }
      }
      if (lane == 0) sfar = ii;
    }
    __syncthreads();
    far = sfar;
  }
}

// ---------------- ball query: 1 block / batch, 1 thread / center ----------------

__global__ __launch_bounds__(256) void ballq_kernel(const float* __restrict__ xyz,
                                                    const float* __restrict__ new_xyz,
                                                    int* __restrict__ idx) {
  int b = blockIdx.x;
  __shared__ float sx[NK], sy[NK], sz[NK];
  int p = threadIdx.x;
  for (int i = p; i < NK; i += 256) {
    sx[i] = xyz[((size_t)b * NK + i) * 3 + 0];
    sy[i] = xyz[((size_t)b * NK + i) * 3 + 1];
    sz[i] = xyz[((size_t)b * NK + i) * 3 + 2];
  }
  __syncthreads();
  float cx = new_xyz[((size_t)b * NPT + p) * 3 + 0];
  float cy = new_xyz[((size_t)b * NPT + p) * 3 + 1];
  float cz = new_xyz[((size_t)b * NPT + p) * 3 + 2];
  float cn = cx * cx + cy * cy + cz * cz;
  int out[NS];
  int cnt = 0, first = 0;
  for (int i = 0; i < NK; i++) {
    float pn  = sx[i] * sx[i] + sy[i] * sy[i] + sz[i] * sz[i];
    float dot = cx * sx[i] + cy * sy[i] + cz * sz[i];
    float d2  = cn + pn - 2.f * dot;  // same expanded form as reference
    if (d2 < RAD * RAD) {
      if (cnt == 0) first = i;
      out[cnt] = i;
      if (++cnt >= NS) break;
    }
  }
#pragma unroll
  for (int j = 0; j < NS; j++)
    idx[(((size_t)b * NPT) + p) * NS + j] = (j < cnt) ? out[j] : first;
}

// ---------------- SA MLP: 8 centers (128 rows) / block, 8 waves, WMMA ----------------
// LDS: Xs 128x296 | Y1 128x136 | W0 128x296 | W1 128x136 | W2 128x136  = 256000 B

__global__ __launch_bounds__(256)
void sa_kernel(const float* __restrict__ xyz, const float* __restrict__ new_xyz,
               const int* __restrict__ ballIdx, const _Float16* __restrict__ featT,
               const _Float16* __restrict__ w0, const float* __restrict__ s0, const float* __restrict__ t0,
               const _Float16* __restrict__ w1, const float* __restrict__ s1, const float* __restrict__ t1,
               const _Float16* __restrict__ w2, const float* __restrict__ s2, const float* __restrict__ t2,
               _Float16* __restrict__ fT) {
  extern __shared__ _Float16 lds[];
  _Float16* Xs = lds;                        // 128 x XSP
  _Float16* Y1 = Xs + 128 * XSP;             // 128 x YS
  _Float16* W0 = Y1 + 128 * YS;              // 128 x XSP
  _Float16* W1 = W0 + 128 * XSP;             // 128 x YS
  _Float16* W2 = W1 + 128 * YS;              // 128 x YS
  _Float16* Y2 = lds;                        // reuse Xs region (dead after layer 0)
  int b  = blockIdx.x >> 5;
  int pg = blockIdx.x & 31;
  int tid = threadIdx.x;
  // stage weights to LDS (bank-conflict-free strides)
  stage_w(w0, W0, 128, K0P, K0P, XSP, tid);
  stage_w(w1, W1, 128, 128, 128, YS, tid);
  stage_w(w2, W2, 128, 128, 128, YS, tid);
  // build X tile: rows = 8 centers x 16 neighbors; cols: [gf 0..255 | gx 256..258 | 0 pad]
  {
    int r = tid >> 1, half = tid & 1;
    int c = r >> 4, s = r & 15;
    int p = pg * 8 + c;
    int g = ballIdx[(((size_t)b * NPT) + p) * NS + s];
    const uint4* src = (const uint4*)(featT + ((size_t)b * NK + g) * NC + half * 128);
    uint4* dst = (uint4*)(Xs + r * XSP + half * 128);
#pragma unroll
    for (int i = 0; i < 16; i++) dst[i] = src[i];
    if (half == 0) {
#pragma unroll
      for (int j = 0; j < 3; j++) {
        float v = (xyz[((size_t)b * NK + g) * 3 + j] -
                   new_xyz[((size_t)b * NPT + p) * 3 + j]) / RAD;
        Xs[r * XSP + 256 + j] = (_Float16)v;
      }
      for (int j = 259; j < K0P; j++) Xs[r * XSP + j] = (_Float16)0.f;
    }
  }
  __syncthreads();
  int lane = tid & 31, w = tid >> 5;
  int rbase = w * 16;  // wave w owns the 16 neighbors of center pg*8+w
  v8f acc[8];
  // layer 0: K = 288
#pragma unroll
  for (int i = 0; i < 8; i++) acc[i] = zero8();
  for (int ks = 0; ks < K0P / 32; ks++) {
    v16h a = load_a(Xs, XSP, rbase, ks * 32, lane);
    v16h bf[8];
#pragma unroll
    for (int ot = 0; ot < 8; ot++) bf[ot] = load_b(W0, XSP, ot * 16, ks * 32, lane);
#pragma unroll
    for (int ot = 0; ot < 8; ot++) acc[ot] = wmma16(a, bf[ot], acc[ot]);
  }
  store_tile_relu(acc, Y1, YS, rbase, s0, t0, lane);
  __syncthreads();
  // layer 1: K = 128
#pragma unroll
  for (int i = 0; i < 8; i++) acc[i] = zero8();
#pragma unroll
  for (int ks = 0; ks < 4; ks++) {
    v16h a = load_a(Y1, YS, rbase, ks * 32, lane);
    v16h bf[8];
#pragma unroll
    for (int ot = 0; ot < 8; ot++) bf[ot] = load_b(W1, YS, ot * 16, ks * 32, lane);
#pragma unroll
    for (int ot = 0; ot < 8; ot++) acc[ot] = wmma16(a, bf[ot], acc[ot]);
  }
  store_tile_relu(acc, Y2, YS, rbase, s1, t1, lane);
  __syncthreads();
  // layer 2 + relu + max over 16 neighbors (in-wave) + store f^T (point-major)
#pragma unroll
  for (int i = 0; i < 8; i++) acc[i] = zero8();
#pragma unroll
  for (int ks = 0; ks < 4; ks++) {
    v16h a = load_a(Y2, YS, rbase, ks * 32, lane);
    v16h bf[8];
#pragma unroll
    for (int ot = 0; ot < 8; ot++) bf[ot] = load_b(W2, YS, ot * 16, ks * 32, lane);
#pragma unroll
    for (int ot = 0; ot < 8; ot++) acc[ot] = wmma16(a, bf[ot], acc[ot]);
  }
  int col = lane & 15;
  int p = pg * 8 + w;
#pragma unroll
  for (int ot = 0; ot < 8; ot++) {
    int o = ot * 16 + col;
    float s = s2[o], t = t2[o];
    float m = 0.f;  // relu outputs >= 0, so 0 is a safe identity
#pragma unroll
    for (int v = 0; v < 8; v++)
      m = fmaxf(m, fmaxf(acc[ot][v] * s + t, 0.f));
    float m2 = __shfl_xor(m, 16, 32);  // combine row halves (same column)
    m = fmaxf(m, m2);
    if (lane < 16)
      fT[(((size_t)b * NPT) + p) * 128 + o] = (_Float16)m;
  }
}

// ---------------- tail FCs: 128 points / block, WMMA ----------------
// LDS: A 128x136 | Bt 128x136 | W1 128x136 | W2 128x136 | W3 112x136 = 169728 B

__global__ __launch_bounds__(256)
void tail_kernel(const _Float16* __restrict__ fT,
                 const _Float16* __restrict__ w1, const float* __restrict__ s1, const float* __restrict__ t1,
                 const _Float16* __restrict__ w2, const float* __restrict__ s2, const float* __restrict__ t2,
                 const _Float16* __restrict__ w3, const float* __restrict__ b3,
                 float* __restrict__ out) {
  extern __shared__ _Float16 lds2[];
  _Float16* A  = lds2;
  _Float16* Bt = A + 128 * YS;
  _Float16* W1 = Bt + 128 * YS;
  _Float16* W2 = W1 + 128 * YS;
  _Float16* W3 = W2 + 128 * YS;
  int blk = blockIdx.x, tid = threadIdx.x;
  stage_w(w1, W1, 128, 128, 128, YS, tid);
  stage_w(w2, W2, 128, 128, 128, YS, tid);
  stage_w(w3, W3, 112, 128, 128, YS, tid);
  {
    int r = tid >> 1, half = tid & 1;
    size_t g = (size_t)blk * 128 + r;
    const uint4* src = (const uint4*)(fT + g * 128 + half * 64);
    uint4* dst = (uint4*)(A + r * YS + half * 64);
#pragma unroll
    for (int i = 0; i < 8; i++) dst[i] = src[i];
  }
  __syncthreads();
  int lane = tid & 31, w = tid >> 5;
  int rbase = w * 16;
  v8f acc[8];
  // h1
#pragma unroll
  for (int i = 0; i < 8; i++) acc[i] = zero8();
#pragma unroll
  for (int ks = 0; ks < 4; ks++) {
    v16h a = load_a(A, YS, rbase, ks * 32, lane);
    v16h bf[8];
#pragma unroll
    for (int ot = 0; ot < 8; ot++) bf[ot] = load_b(W1, YS, ot * 16, ks * 32, lane);
#pragma unroll
    for (int ot = 0; ot < 8; ot++) acc[ot] = wmma16(a, bf[ot], acc[ot]);
  }
  store_tile_relu(acc, Bt, YS, rbase, s1, t1, lane);
  __syncthreads();
  // h2
#pragma unroll
  for (int i = 0; i < 8; i++) acc[i] = zero8();
#pragma unroll
  for (int ks = 0; ks < 4; ks++) {
    v16h a = load_a(Bt, YS, rbase, ks * 32, lane);
    v16h bf[8];
#pragma unroll
    for (int ot = 0; ot < 8; ot++) bf[ot] = load_b(W2, YS, ot * 16, ks * 32, lane);
#pragma unroll
    for (int ot = 0; ot < 8; ot++) acc[ot] = wmma16(a, bf[ot], acc[ot]);
  }
  store_tile_relu(acc, A, YS, rbase, s2, t2, lane);
  __syncthreads();
  // head: 97 (padded 112) outputs
  v8f oacc[7];
#pragma unroll
  for (int i = 0; i < 7; i++) oacc[i] = zero8();
#pragma unroll
  for (int ks = 0; ks < 4; ks++) {
    v16h a = load_a(A, YS, rbase, ks * 32, lane);
    v16h bf[7];
#pragma unroll
    for (int ot = 0; ot < 7; ot++) bf[ot] = load_b(W3, YS, ot * 16, ks * 32, lane);
#pragma unroll
    for (int ot = 0; ot < 7; ot++) oacc[ot] = wmma16(a, bf[ot], oacc[ot]);
  }
  int col  = lane & 15;
  int rofs = (lane & 16) ? 8 : 0;
#pragma unroll
  for (int ot = 0; ot < 7; ot++) {
    int o = ot * 16 + col;
    if (o < OUTCH) {
      float bias = b3[o];
#pragma unroll
      for (int v = 0; v < 8; v++) {
        int g = blk * 128 + rbase + rofs + v;
        int bb = g >> 8, p = g & 255;
        out[(((size_t)bb * OUTCH) + o) * NPT + p] = oacc[ot][v] + bias;
      }
    }
  }
}

// ---------------- launcher ----------------

extern "C" void kernel_launch(void* const* d_in, const int* in_sizes, int n_in,
                              void* d_out, int out_size, void* d_ws, size_t ws_size,
                              hipStream_t stream) {
  (void)in_sizes; (void)n_in; (void)out_size; (void)ws_size;
  const float* xyz      = (const float*)d_in[0];
  const float* features = (const float*)d_in[1];
  const float* sa_w0    = (const float*)d_in[2];
  const float* sa_s0    = (const float*)d_in[3];
  const float* sa_t0    = (const float*)d_in[4];
  const float* sa_w1    = (const float*)d_in[5];
  const float* sa_s1    = (const float*)d_in[6];
  const float* sa_t1    = (const float*)d_in[7];
  const float* sa_w2    = (const float*)d_in[8];
  const float* sa_s2    = (const float*)d_in[9];
  const float* sa_t2    = (const float*)d_in[10];
  const float* w1       = (const float*)d_in[11];
  const float* s1       = (const float*)d_in[12];
  const float* t1       = (const float*)d_in[13];
  const float* w2       = (const float*)d_in[14];
  const float* s2       = (const float*)d_in[15];
  const float* t2       = (const float*)d_in[16];
  const float* w3       = (const float*)d_in[17];
  const float* b3       = (const float*)d_in[18];

  char* ws = (char*)d_ws;
  size_t off = 0;
  auto take = [&](size_t bytes) -> char* {
    char* p = ws + off;
    off += (bytes + 255) & ~(size_t)255;
    return p;
  };
  float*     new_xyz = (float*)take((size_t)BATCH * NPT * 3 * 4);
  int*       ballIdx = (int*)take((size_t)BATCH * NPT * NS * 4);
  _Float16*  featT   = (_Float16*)take((size_t)BATCH * NK * NC * 2);
  _Float16*  w0h     = (_Float16*)take((size_t)128 * K0P * 2);
  _Float16*  w1sa    = (_Float16*)take((size_t)128 * 128 * 2);
  _Float16*  w2sa    = (_Float16*)take((size_t)128 * 128 * 2);
  _Float16*  w1h     = (_Float16*)take((size_t)128 * 128 * 2);
  _Float16*  w2h     = (_Float16*)take((size_t)128 * 128 * 2);
  _Float16*  w3h     = (_Float16*)take((size_t)112 * 128 * 2);
  _Float16*  fT      = (_Float16*)take((size_t)BATCH * NPT * 128 * 2);

  cvt_w0_kernel<<<(128 * K0P + 255) / 256, 256, 0, stream>>>(sa_w0, w0h);
  cvt128_kernel<<<(128 * 128 + 255) / 256, 256, 0, stream>>>(sa_w1, w1sa);
  cvt128_kernel<<<(128 * 128 + 255) / 256, 256, 0, stream>>>(sa_w2, w2sa);
  cvt128_kernel<<<(128 * 128 + 255) / 256, 256, 0, stream>>>(w1, w1h);
  cvt128_kernel<<<(128 * 128 + 255) / 256, 256, 0, stream>>>(w2, w2h);
  cvt_w3_kernel<<<(112 * 128 + 255) / 256, 256, 0, stream>>>(w3, w3h);

  feat_transpose_kernel<<<dim3(NK / 32, NC / 32, BATCH), dim3(32, 8, 1), 0, stream>>>(features, featT);
  fps_kernel<<<BATCH, 1024, 0, stream>>>(xyz, new_xyz);
  ballq_kernel<<<BATCH, 256, 0, stream>>>(xyz, new_xyz, ballIdx);

  size_t sa_lds   = (size_t)(2 * 128 * XSP + 3 * 128 * YS) * 2;        // 256000 B
  size_t tail_lds = (size_t)(4 * 128 * YS + 112 * YS) * 2;             // 169728 B
  sa_kernel<<<BATCH * (NPT / 8), 256, sa_lds, stream>>>(
      xyz, new_xyz, ballIdx, featT,
      w0h, sa_s0, sa_t0, w1sa, sa_s1, sa_t1, w2sa, sa_s2, sa_t2, fT);
  tail_kernel<<<(BATCH * NPT) / 128, 256, tail_lds, stream>>>(
      fT, w1h, s1, t1, w2h, s2, t2, w3h, b3, (float*)d_out);
}
